// MultiHeadAttention_15221364097501
// MI455X (gfx1250) — compile-verified, matrix operand order checked by reading, not run on hardware
//
#include <hip/hip_runtime.h>
#include <hip/hip_bf16.h>

typedef _Float16 h16;
typedef __attribute__((ext_vector_type(16))) _Float16 v16h;
typedef __attribute__((ext_vector_type(8)))  _Float16 v8h;
typedef __attribute__((ext_vector_type(8)))  float    v8f;

constexpr int B  = 4;
constexpr int S  = 2048;
constexpr int D  = 1024;
constexpr int H  = 16;
constexpr int DK = 64;
constexpr int M  = B * S;              // 8192 rows
constexpr size_t XE = (size_t)M * D;   // 8,388,608 elements
constexpr size_t WE = (size_t)D * D;   // 1,048,576 elements

// LDS score strip row stride (floats). 2052 % 64 == 4 -> conflict-free
// phase-1 stores, <=2-way conflict phase-3 frag reads; 2052*4 % 16 == 0 so
// 16B vector loads stay aligned.
constexpr int SLD = S + 4;             // 2052

// ---------------------------------------------------------------------------
// WMMA helpers (CDNA5 gfx1250, wave32).
// A frag (16x32 f16): lane l: m=l&15, hi=l>>4; halves 0..7 <- K=8*hi..8*hi+7,
// halves 8..15 <- K=16+8*hi..16+8*hi+7  (per CDNA5 ISA A-matrix table).
// B frag (32x16 f16) mirrors A with lane = N column; columns of B are rows of
// the N-major operand buffer, so both frags are contiguous 16B loads.
// C/D (16x16 f32): lane l: n=l&15, vgpr i -> row i + 8*(l>>4).
// ---------------------------------------------------------------------------
union V16HU { v16h v; v8h h[2]; };

__device__ __forceinline__ v16h ld_frag(const h16* __restrict__ p) {
  V16HU u;
  u.h[0] = *(const v8h*)(p);
  u.h[1] = *(const v8h*)(p + 16);
  return u.v;
}

__device__ __forceinline__ v16h ld_frag_f32(const float* p) {
  V16HU u;
#pragma unroll
  for (int i = 0; i < 8; ++i) u.h[0][i] = (h16)p[i];
#pragma unroll
  for (int i = 0; i < 8; ++i) u.h[1][i] = (h16)p[16 + i];
  return u.v;
}

__device__ __forceinline__ v8f wmma16(v16h a, v16h b, v8f c) {
  return __builtin_amdgcn_wmma_f32_16x16x32_f16(false, a, false, b,
                                                (short)0, c, false, false);
}

// ---------------------------------------------------------------------------
// Kernel 1: fp32 -> f16 conversion of x and the four weight matrices.
// ---------------------------------------------------------------------------
__global__ __launch_bounds__(256) void cvt_kernel(
    const float* __restrict__ x,
    const float* __restrict__ Wq, const float* __restrict__ Wk,
    const float* __restrict__ Wv, const float* __restrict__ Wo,
    h16* __restrict__ xh, h16* __restrict__ Wqh, h16* __restrict__ Wkh,
    h16* __restrict__ Wvh, h16* __restrict__ Woh) {
  size_t idx = (size_t)blockIdx.x * 256 + threadIdx.x;
  if (idx < XE) {
    xh[idx] = (h16)x[idx];
  } else {
    size_t w = idx - XE;
    int which = (int)(w >> 20);          // WE == 2^20
    size_t off = w & (WE - 1);
    const float* src = (which == 0) ? Wq : (which == 1) ? Wk
                     : (which == 2) ? Wv : Wo;
    h16* dst = (which == 0) ? Wqh : (which == 1) ? Wkh
             : (which == 2) ? Wvh : Woh;
    dst[off] = (h16)src[off];
  }
}

// ---------------------------------------------------------------------------
// Kernel 2: fused Q/K/V projection GEMM.  O = x @ W^T + b  (M=8192, N=D=1024,
// K=D=1024).  blockIdx.z selects Q/K/V.  Workgroup = 256 threads (8 waves),
// block tile 128(M) x 64(N), each wave a 32x32 tile (4 wmma per K-step).
// Q,K stored as [B,H,S,DK] f16; V stored transposed [B,H,DK,S] f16.
// ---------------------------------------------------------------------------
__global__ __launch_bounds__(256) void qkv_kernel(
    const h16* __restrict__ xh,
    const h16* __restrict__ Wqh, const h16* __restrict__ Wkh,
    const h16* __restrict__ Wvh,
    const float* __restrict__ bq, const float* __restrict__ bk,
    const float* __restrict__ bv,
    h16* __restrict__ Qh, h16* __restrict__ Kh, h16* __restrict__ VTh) {
  const int z    = blockIdx.z;
  const h16* Wh  = (z == 0) ? Wqh : (z == 1) ? Wkh : Wvh;
  const float* bias = (z == 0) ? bq : (z == 1) ? bk : bv;
  const int wave = threadIdx.x >> 5;
  const int lane = threadIdx.x & 31;
  const int l15  = lane & 15;
  const int hi   = lane >> 4;
  const int mBase = blockIdx.x * 128 + (wave >> 1) * 32;
  const int nBase = blockIdx.y * 64 + (wave & 1) * 32;

  v8f acc[2][2] = {};
  const h16* arow = xh + (size_t)(mBase + l15) * D + 8 * hi;
  const h16* brow = Wh + (size_t)(nBase + l15) * D + 8 * hi;

  for (int k0 = 0; k0 < D; k0 += 32) {
    v16h a0 = ld_frag(arow + k0);
    v16h a1 = ld_frag(arow + k0 + (size_t)16 * D);
    v16h b0 = ld_frag(brow + k0);
    v16h b1 = ld_frag(brow + k0 + (size_t)16 * D);
    acc[0][0] = wmma16(a0, b0, acc[0][0]);
    acc[0][1] = wmma16(a0, b1, acc[0][1]);
    acc[1][0] = wmma16(a1, b0, acc[1][0]);
    acc[1][1] = wmma16(a1, b1, acc[1][1]);
  }

#pragma unroll
  for (int ti = 0; ti < 2; ++ti)
#pragma unroll
    for (int tj = 0; tj < 2; ++tj) {
      int n = nBase + tj * 16 + l15;
      float bval = bias[n];
      int head = n >> 6;     // n / DK
      int dk   = n & 63;     // n % DK
#pragma unroll
      for (int i = 0; i < 8; ++i) {
        int mg = mBase + ti * 16 + 8 * hi + i;
        int bb = mg >> 11;   // mg / S
        int s  = mg & 2047;  // mg % S
        float val = acc[ti][tj][i] + bval;
        if (z == 2) {
          VTh[(size_t)((bb * H + head) * DK + dk) * S + s] = (h16)val;
        } else {
          h16* dst = (z == 0) ? Qh : Kh;
          dst[(size_t)((bb * H + head) * S + s) * DK + dk] = (h16)val;
        }
      }
    }
}

// ---------------------------------------------------------------------------
// Kernel 3: attention for one (b, h, 32-query strip).
//   Phase 1: scores = scale * Q Kt  -> LDS strip [32][SLD] fp32.
//   Phase 2: exact row softmax (wave-per-4-rows, lane-contiguous sweeps,
//            __shfl_xor reductions); nontemporal write of normalized attn.
//   Phase 3: ctx = P @ V (unnormalized e, rescaled by 1/rowsum), f16 out.
// Dynamic LDS: 32*SLD*4 + 32*4 = 262784 bytes (< 320KB WGP LDS).
// ---------------------------------------------------------------------------
constexpr int SMEM_BYTES = 32 * SLD * 4 + 32 * 4;

__global__ __launch_bounds__(256) void attn_kernel(
    const h16* __restrict__ Qh, const h16* __restrict__ Kh,
    const h16* __restrict__ VTh,
    float* __restrict__ attnOut, h16* __restrict__ CTXh) {
  extern __shared__ float smem[];
  float* Sc     = smem;                  // [32][SLD]
  float* rowsum = Sc + 32 * SLD;         // [32]

  const int b  = blockIdx.z;
  const int h  = blockIdx.y;
  const int q0 = blockIdx.x * 32;
  const int wave = threadIdx.x >> 5;
  const int lane = threadIdx.x & 31;
  const int l15  = lane & 15;
  const int hi   = lane >> 4;
  const float scale = 0.125f;            // 1/sqrt(DK)

  const h16* Qb  = Qh  + ((size_t)(b * H + h) * S + q0) * DK;
  const h16* Kb  = Kh  + (size_t)(b * H + h) * S * DK;
  const h16* VTb = VTh + (size_t)(b * H + h) * DK * S;

  // --- Phase 1: scores into LDS ------------------------------------------
  v16h aq[2][2];
#pragma unroll
  for (int mi = 0; mi < 2; ++mi)
#pragma unroll
    for (int ks = 0; ks < 2; ++ks)
      aq[mi][ks] = ld_frag(Qb + (size_t)(mi * 16 + l15) * DK + ks * 32 + 8 * hi);

  for (int kb = wave; kb < S / 32; kb += 8) {
    v8f s00 = {}, s01 = {}, s10 = {}, s11 = {};
#pragma unroll
    for (int ks = 0; ks < 2; ++ks) {
      const h16* kp = Kb + (size_t)(kb * 32 + l15) * DK + ks * 32 + 8 * hi;
      v16h b0 = ld_frag(kp);
      v16h b1 = ld_frag(kp + (size_t)16 * DK);
      s00 = wmma16(aq[0][ks], b0, s00);
      s01 = wmma16(aq[0][ks], b1, s01);
      s10 = wmma16(aq[1][ks], b0, s10);
      s11 = wmma16(aq[1][ks], b1, s11);
    }
#pragma unroll
    for (int i = 0; i < 8; ++i) {
      int col = kb * 32 + l15;
      Sc[(8 * hi + i) * SLD + col]           = s00[i] * scale;
      Sc[(8 * hi + i) * SLD + col + 16]      = s01[i] * scale;
      Sc[(16 + 8 * hi + i) * SLD + col]      = s10[i] * scale;
      Sc[(16 + 8 * hi + i) * SLD + col + 16] = s11[i] * scale;
    }
  }
  __syncthreads();

  // --- Phase 2: exact softmax + coalesced attn writeback -----------------
  // Wave w owns rows 4w..4w+3; lanes stride 32 across the 2048 columns:
  // LDS accesses are conflict-free and global stores are 128B-contiguous.
#pragma unroll
  for (int rr = 0; rr < 4; ++rr) {
    const int r = wave * 4 + rr;
    float* row = Sc + r * SLD;

    float mx = -3.4e38f;
    for (int j = lane; j < S; j += 32) mx = fmaxf(mx, row[j]);
#pragma unroll
    for (int o = 16; o > 0; o >>= 1) mx = fmaxf(mx, __shfl_xor(mx, o, 32));

    float sum = 0.0f;
    for (int j = lane; j < S; j += 32) {
      float e = __expf(row[j] - mx);
      row[j] = e;                        // keep unnormalized e in LDS
      sum += e;
    }
#pragma unroll
    for (int o = 16; o > 0; o >>= 1) sum += __shfl_xor(sum, o, 32);
    if (lane == 0) rowsum[r] = sum;

    const float inv = 1.0f / sum;
    float* arow = attnOut + ((size_t)(b * H + h) * S + q0 + r) * S;
    for (int j = lane; j < S; j += 32)
      __builtin_nontemporal_store(row[j] * inv, arow + j);
  }
  __syncthreads();

  // --- Phase 3: ctx = P @ V (rescale rows by 1/rowsum) -------------------
  {
    const int mi = wave >> 2;            // 0..1 : query sub-tile
    const int nj = wave & 3;             // 0..3 : dk sub-tile
    v8f acc = {};
    const float* prow = Sc + (mi * 16 + l15) * SLD + 8 * hi;
    const h16* vrow = VTb + (size_t)(nj * 16 + l15) * S + 8 * hi;
    for (int k0 = 0; k0 < S; k0 += 32) {
      v16h a = ld_frag_f32(prow + k0);
      v16h v = ld_frag(vrow + k0);
      acc = wmma16(a, v, acc);
    }
#pragma unroll
    for (int i = 0; i < 8; ++i) {
      int rr = mi * 16 + 8 * hi + i;
      float val = acc[i] * (1.0f / rowsum[rr]);
      int sg = q0 + rr;
      CTXh[(size_t)(b * S + sg) * D + h * DK + nj * 16 + l15] = (h16)val;
    }
  }
}

// ---------------------------------------------------------------------------
// Kernel 4: output projection.  out = ctx @ Wo^T + bo  (fp32 out).
// ---------------------------------------------------------------------------
__global__ __launch_bounds__(256) void oproj_kernel(
    const h16* __restrict__ CTXh, const h16* __restrict__ Woh,
    const float* __restrict__ bo, float* __restrict__ out) {
  const int wave = threadIdx.x >> 5;
  const int lane = threadIdx.x & 31;
  const int l15  = lane & 15;
  const int hi   = lane >> 4;
  const int mBase = blockIdx.x * 128 + (wave >> 1) * 32;
  const int nBase = blockIdx.y * 64 + (wave & 1) * 32;

  v8f acc[2][2] = {};
  const h16* arow = CTXh + (size_t)(mBase + l15) * D + 8 * hi;
  const h16* brow = Woh + (size_t)(nBase + l15) * D + 8 * hi;

  for (int k0 = 0; k0 < D; k0 += 32) {
    v16h a0 = ld_frag(arow + k0);
    v16h a1 = ld_frag(arow + k0 + (size_t)16 * D);
    v16h b0 = ld_frag(brow + k0);
    v16h b1 = ld_frag(brow + k0 + (size_t)16 * D);
    acc[0][0] = wmma16(a0, b0, acc[0][0]);
    acc[0][1] = wmma16(a0, b1, acc[0][1]);
    acc[1][0] = wmma16(a1, b0, acc[1][0]);
    acc[1][1] = wmma16(a1, b1, acc[1][1]);
  }

#pragma unroll
  for (int ti = 0; ti < 2; ++ti)
#pragma unroll
    for (int tj = 0; tj < 2; ++tj) {
      int n = nBase + tj * 16 + l15;
      float bval = bo[n];
#pragma unroll
      for (int i = 0; i < 8; ++i) {
        int mg = mBase + ti * 16 + 8 * hi + i;
        out[(size_t)mg * D + n] = acc[ti][tj][i] + bval;
      }
    }
}

// ---------------------------------------------------------------------------
// Host-side launcher.
// Workspace layout (f16): xh | Wqh | Wkh | Wvh | Woh | Qh | Kh | VTh | CTXh
// Total = 5*XE + 4*WE halves = 92,274,688 bytes.
// ---------------------------------------------------------------------------
extern "C" void kernel_launch(void* const* d_in, const int* in_sizes, int n_in,
                              void* d_out, int out_size, void* d_ws,
                              size_t ws_size, hipStream_t stream) {
  (void)in_sizes; (void)n_in; (void)out_size; (void)ws_size;

  const float* x  = (const float*)d_in[0];
  const float* Wq = (const float*)d_in[1];
  const float* bq = (const float*)d_in[2];
  const float* Wk = (const float*)d_in[3];
  const float* bk = (const float*)d_in[4];
  const float* Wv = (const float*)d_in[5];
  const float* bv = (const float*)d_in[6];
  const float* Wo = (const float*)d_in[7];
  const float* bo = (const float*)d_in[8];

  float* out  = (float*)d_out;                 // [B,S,D]
  float* attn = out + (size_t)B * S * D;       // [B,H,S,S]

  h16* xh   = (h16*)d_ws;
  h16* Wqh  = xh + XE;
  h16* Wkh  = Wqh + WE;
  h16* Wvh  = Wkh + WE;
  h16* Woh  = Wvh + WE;
  h16* Qh   = Woh + WE;
  h16* Kh   = Qh + XE;
  h16* VTh  = Kh + XE;
  h16* CTXh = VTh + XE;

  // 1) fp32 -> f16 conversions
  {
    size_t total = XE + 4 * WE;               // 12,582,912
    dim3 grid((unsigned)((total + 255) / 256));
    cvt_kernel<<<grid, 256, 0, stream>>>(x, Wq, Wk, Wv, Wo,
                                         xh, Wqh, Wkh, Wvh, Woh);
  }

  // 2) Q/K/V projections (z selects which)
  qkv_kernel<<<dim3(M / 128, D / 64, 3), 256, 0, stream>>>(
      xh, Wqh, Wkh, Wvh, bq, bk, bv, Qh, Kh, VTh);

  // 3) attention (scores + softmax + attn writeback + P@V)
  hipFuncSetAttribute(reinterpret_cast<const void*>(attn_kernel),
                      hipFuncAttributeMaxDynamicSharedMemorySize, SMEM_BYTES);
  attn_kernel<<<dim3(S / 32, H, B), 256, SMEM_BYTES, stream>>>(
      Qh, Kh, VTh, attn, CTXh);

  // 4) output projection
  oproj_kernel<<<dim3(M / 128, D / 64), 256, 0, stream>>>(CTXh, Woh, bo, out);
}
